// DiscreteMamba2_83476984365502
// MI455X (gfx1250) — matrix operand order repaired and checked by reading, not compiled, not run on hardware
//
#include <hip/hip_runtime.h>
#include <hip/hip_bf16.h>

#define D_MODEL 2048
#define D_INNER 4096
#define D_STATE 64
#define NQK 8
#define NVH 64
#define CHUNKL 128
#define KSZ 4
#define CONV_DIM (D_INNER + 2*NQK*D_STATE)   /* 5120 */
#define PROJ_DIM (CONV_DIM + D_INNER + NVH)  /* 9280 */
#define BATCH 2
#define SEQ 4096
#define TOK (BATCH*SEQ)   /* 8192 */
#define NCH (SEQ/CHUNKL)  /* 32 chunks per batch */
#define HP 64             /* v-head dim p */
#define NST 64            /* state dim n */

typedef __bf16 bf16;
typedef __attribute__((ext_vector_type(16))) __bf16 v16bf;
typedef __attribute__((ext_vector_type(8)))  __bf16 v8bf;
typedef __attribute__((ext_vector_type(8)))  float  v8f;

static __device__ __forceinline__ v8f zero_v8f() {
  v8f z;
#pragma unroll
  for (int i = 0; i < 8; ++i) z[i] = 0.f;
  return z;
}

static __device__ __forceinline__ v8f wmma_bf16(v16bf a, v16bf b, v8f c) {
  // D = A(16x32 bf16) * B(32x16 bf16) + C(16x16 f32)
  return __builtin_amdgcn_wmma_f32_16x16x32_bf16(false, a, false, b, (short)0, c,
                                                 false, false);
}

// Combine two 8-elem halves (k-lo / k-hi) into one A fragment register set.
static __device__ __forceinline__ v16bf combine16(v8bf lo, v8bf hi) {
  v16bf r;
#pragma unroll
  for (int i = 0; i < 8; ++i) { r[i] = lo[i]; r[i + 8] = hi[i]; }
  return r;
}

// A fragment: 16x32 tile from row-major [m][k] storage with leading dim ld.
// Lane l holds row m=l&15, 16 contiguous k at (l>>4)*16.
static __device__ __forceinline__ v16bf load_a_frag(const bf16* p, int ld, int lane) {
  const bf16* q = p + (lane & 15) * ld + ((lane >> 4) << 4);
  return *(const v16bf*)q;
}

// B fragment from B-transposed storage [n][k] (leading dim ld):
// lane l holds column n=l&15, 16 contiguous k at (l>>4)*16.
static __device__ __forceinline__ v16bf load_bt_frag(const bf16* p, int ld, int lane) {
  const bf16* q = p + (lane & 15) * ld + ((lane >> 4) << 4);
  return *(const v16bf*)q;
}

// ---------------- elementwise kernels ----------------

__global__ void cvt_f32_bf16(const float* __restrict__ src, bf16* __restrict__ dst, int n) {
  int i = blockIdx.x * 256 + threadIdx.x;
  if (i < n) dst[i] = (bf16)src[i];
}

// Depthwise causal conv (K=4) + bias + silu over CONV_DIM channels -> bf16
__global__ void conv_silu_kernel(const float* __restrict__ xbcza,
                                 const float* __restrict__ w,
                                 const float* __restrict__ bias,
                                 bf16* __restrict__ xbc) {
  int idx = blockIdx.x * 256 + threadIdx.x;
  int ch = idx % CONV_DIM;
  int t  = idx / CONV_DIM;
  if (t >= TOK) return;
  int l = t & (SEQ - 1);
  float acc = bias[ch];
#pragma unroll
  for (int k = 0; k < KSZ; ++k) {
    int ls = l - (KSZ - 1) + k;
    if (ls >= 0)
      acc += w[ch * KSZ + k] * xbcza[(size_t)(t - (KSZ - 1) + k) * PROJ_DIM + ch];
  }
  float s = acc / (1.f + __expf(-acc));
  xbc[(size_t)t * CONV_DIM + ch] = (bf16)s;
}

__global__ void dt_kernel(const float* __restrict__ xbcza, float* __restrict__ dt) {
  int idx = blockIdx.x * 256 + threadIdx.x;
  if (idx >= TOK * NVH) return;
  int t = idx / NVH, h = idx % NVH;
  float a = xbcza[(size_t)t * PROJ_DIM + CONV_DIM + D_INNER + h];
  dt[idx] = (a > 20.f) ? a : log1pf(__expf(a));
}

// ---------------- WMMA GEMM: C[M][N] = A[M][K] * W[N][K]^T ----------------
// one wave per block; wave tile 128(M) x 32(N); K%32==0; M%128==0; N%32==0.
// Per-lane fragment pointers advance by +32 elements per k-step so the inner
// loop is pure {load_b128 x2, wmma} with immediate offsets.
__global__ void __launch_bounds__(32)
gemm_bf16_wmma(const bf16* __restrict__ A, const bf16* __restrict__ W,
               float* __restrict__ C, int M, int N, int K) {
  int lane = threadIdx.x & 31;
  int row0 = blockIdx.x * 128;
  int col0 = blockIdx.y * 32;
  int m = lane & 15, half = lane >> 4;

  // lane-private fragment bases
  const bf16* pa = A + (size_t)(row0 + m) * K + half * 16;  // A frag i at pa + (16*i)*K
  const bf16* pb = W + (size_t)(col0 + m) * K + half * 16;  // B frag j at pb + (16*j)*K

  v8f acc[8][2];
#pragma unroll
  for (int i = 0; i < 8; ++i)
#pragma unroll
    for (int j = 0; j < 2; ++j) acc[i][j] = zero_v8f();

  const size_t strideA = (size_t)16 * K;  // between 16-row sub-tiles
  for (int k0 = 0; k0 < K; k0 += 32) {
    v16bf b0 = *(const v16bf*)pb;
    v16bf b1 = *(const v16bf*)(pb + strideA);
    __builtin_prefetch(pb + 256, 0, 3);  // WGP-scope temporal prefetch
#pragma unroll
    for (int i = 0; i < 8; ++i) {
      const bf16* qa = pa + (size_t)i * strideA;
      v16bf a = *(const v16bf*)qa;
      acc[i][0] = wmma_bf16(a, b0, acc[i][0]);
      acc[i][1] = wmma_bf16(a, b1, acc[i][1]);
    }
    __builtin_prefetch(pa + 256, 0, 3);  // WGP-scope temporal prefetch
    pa += 32;
    pb += 32;
  }

  int rsub = (lane >> 4) * 8, csub = lane & 15;
#pragma unroll
  for (int i = 0; i < 8; ++i)
#pragma unroll
    for (int j = 0; j < 2; ++j)
#pragma unroll
      for (int r = 0; r < 8; ++r) {
        int row = row0 + 16 * i + rsub + r;
        int col = col0 + 16 * j + csub;
        C[(size_t)row * N + col] = acc[i][j][r];
      }
}

// ---------------- SSD intra-chunk: Y_diag + chunk states ----------------
// one block per (b, chunk, head); 256 threads = 8 waves
__global__ void __launch_bounds__(256)
ssd_chunk_kernel(const bf16* __restrict__ xbc, const float* __restrict__ dt,
                 float* __restrict__ Yg, float* __restrict__ states,
                 float* __restrict__ Acg) {
  __shared__ __align__(32) bf16 Cs[CHUNKL * NST];   // Cm [s][n]
  __shared__ __align__(32) bf16 Bs[CHUNKL * NST];   // Bm [s][n]
  __shared__ __align__(32) bf16 BT[NST * CHUNKL];   // Bm^T [n][s]
  __shared__ __align__(32) bf16 XT[HP * CHUNKL];    // X^T  [p][s]
  __shared__ __align__(32) bf16 XdT[HP * CHUNKL];   // (decay*X)^T [p][s]
  __shared__ __align__(32) bf16 Mm[CHUNKL * CHUNKL];// masked G, bf16
  __shared__ float Ac[CHUNKL];
  __shared__ float dts[CHUNKL];

  int bch = blockIdx.x;
  int h = bch & 63;
  int cc = (bch >> 6) & 31;
  int b = bch >> 11;
  int g = h >> 3;
  int t0 = b * SEQ + cc * CHUNKL;
  int tid = threadIdx.x;

  if (tid < CHUNKL) dts[tid] = dt[(size_t)(t0 + tid) * NVH + h];
  __syncthreads();
  if (tid == 0) {
    float run = 0.f;
    for (int s = 0; s < CHUNKL; ++s) { run -= dts[s]; Ac[s] = run; }
  }
  __syncthreads();
  float Alast = Ac[CHUNKL - 1];
  if (tid < CHUNKL) Acg[(size_t)bch * CHUNKL + tid] = Ac[tid];

  for (int i = tid; i < CHUNKL * NST; i += 256) {
    int s = i >> 6, j = i & 63;
    size_t base = (size_t)(t0 + s) * CONV_DIM;
    bf16 bv = xbc[base + D_INNER + g * 64 + j];
    bf16 cv = xbc[base + D_INNER + NQK * D_STATE + g * 64 + j];
    bf16 xv = xbc[base + h * 64 + j];
    Bs[s * 64 + j] = bv;
    BT[j * CHUNKL + s] = bv;
    Cs[s * 64 + j] = cv;
    XT[j * CHUNKL + s] = xv;
    XdT[j * CHUNKL + s] = (bf16)((float)xv * __expf(Alast - Ac[s]));
  }
  __syncthreads();

  int w = tid >> 5, lane = tid & 31;
  int rsub = (lane >> 4) * 8, csub = lane & 15;

  // G = Cm @ Bm^T (128x128, K=64), mask with exp(segsum), store bf16 to Mm
  {
    v8f acc[8];
#pragma unroll
    for (int n = 0; n < 8; ++n) acc[n] = zero_v8f();
    for (int k0 = 0; k0 < NST; k0 += 32) {
      v16bf a = load_a_frag(Cs + (w * 16) * NST + k0, NST, lane);
#pragma unroll
      for (int n = 0; n < 8; ++n) {
        v16bf bb = load_bt_frag(Bs + (n * 16) * NST + k0, NST, lane);
        acc[n] = wmma_bf16(a, bb, acc[n]);
      }
    }
#pragma unroll
    for (int n = 0; n < 8; ++n)
#pragma unroll
      for (int r = 0; r < 8; ++r) {
        int i = w * 16 + rsub + r;
        int j = n * 16 + csub;
        float v = (i >= j) ? acc[n][r] * __expf(Ac[i] - Ac[j]) : 0.f;
        Mm[i * CHUNKL + j] = (bf16)v;
      }
  }
  __syncthreads();

  // Y_diag = Mm(128x128) @ X(128x64)  [B^T storage = XT]
  {
    v8f acc[4];
#pragma unroll
    for (int n = 0; n < 4; ++n) acc[n] = zero_v8f();
    for (int k0 = 0; k0 < CHUNKL; k0 += 32) {
      v16bf a = load_a_frag(Mm + (w * 16) * CHUNKL + k0, CHUNKL, lane);
#pragma unroll
      for (int n = 0; n < 4; ++n) {
        v16bf bb = load_bt_frag(XT + (n * 16) * CHUNKL + k0, CHUNKL, lane);
        acc[n] = wmma_bf16(a, bb, acc[n]);
      }
    }
#pragma unroll
    for (int n = 0; n < 4; ++n)
#pragma unroll
      for (int r = 0; r < 8; ++r) {
        int i = w * 16 + rsub + r;
        int p = n * 16 + csub;
        Yg[(size_t)(t0 + i) * D_INNER + h * 64 + p] = acc[n][r];
      }
  }

  // states[p][n] = sum_s XdT[p][s] * Bm[s][n]  (64x64, K=128) [B^T storage = BT]
  {
    int pr = (w >> 1) * 16;
    int nc = (w & 1) * 32;
    v8f acc[2];
#pragma unroll
    for (int n = 0; n < 2; ++n) acc[n] = zero_v8f();
    for (int k0 = 0; k0 < CHUNKL; k0 += 32) {
      v16bf a = load_a_frag(XdT + pr * CHUNKL + k0, CHUNKL, lane);
#pragma unroll
      for (int n = 0; n < 2; ++n) {
        v16bf bb = load_bt_frag(BT + (nc + n * 16) * CHUNKL + k0, CHUNKL, lane);
        acc[n] = wmma_bf16(a, bb, acc[n]);
      }
    }
#pragma unroll
    for (int n = 0; n < 2; ++n)
#pragma unroll
      for (int r = 0; r < 8; ++r) {
        int p = pr + rsub + r;
        int nn = nc + n * 16 + csub;
        states[(size_t)bch * (HP * NST) + p * NST + nn] = acc[n][r];
      }
  }
}

// ---------------- inter-chunk recurrence (sequential over 32 chunks) ----------------
// N[0]=0 ; N[c] = exp(chunk_sum[c-1]) * N[c-1] + S[c-1] ; states_used[c] = N[c]
__global__ void __launch_bounds__(256)
chunk_scan_kernel(const float* __restrict__ states, const float* __restrict__ Acg,
                  float* __restrict__ states_used) {
  int bh = blockIdx.x;  // b*64 + h
  int b = bh >> 6, h = bh & 63;
  int tid = threadIdx.x;
  float prev[16];
#pragma unroll
  for (int i = 0; i < 16; ++i) prev[i] = 0.f;
  {
    int bch0 = b * (NCH * NVH) + h;
#pragma unroll
    for (int i = 0; i < 16; ++i)
      states_used[(size_t)bch0 * (HP * NST) + tid + i * 256] = 0.f;
  }
  for (int c = 1; c < NCH; ++c) {
    int bchp = b * (NCH * NVH) + (c - 1) * NVH + h;
    int bchc = b * (NCH * NVH) + c * NVH + h;
    float dec = __expf(Acg[(size_t)bchp * CHUNKL + (CHUNKL - 1)]);
#pragma unroll
    for (int i = 0; i < 16; ++i) {
      float raw = states[(size_t)bchp * (HP * NST) + tid + i * 256];
      prev[i] = prev[i] * dec + raw;
      states_used[(size_t)bchc * (HP * NST) + tid + i * 256] = prev[i];
    }
  }
}

// ---------------- Y_off = (Cm @ states^T) * exp(A_cum), accumulate into Yg ----------------
__global__ void __launch_bounds__(256)
y_off_kernel(const bf16* __restrict__ xbc, const float* __restrict__ states_used,
             const float* __restrict__ Acg, float* __restrict__ Yg) {
  __shared__ __align__(32) bf16 Cs[CHUNKL * NST];
  __shared__ __align__(32) bf16 stT[HP * NST];   // states [p][n] == B^T storage
  __shared__ float Ac[CHUNKL];
  int bch = blockIdx.x;
  int h = bch & 63, cc = (bch >> 6) & 31, b = bch >> 11, g = h >> 3;
  int t0 = b * SEQ + cc * CHUNKL;
  int tid = threadIdx.x;
  if (tid < CHUNKL) Ac[tid] = Acg[(size_t)bch * CHUNKL + tid];
  for (int i = tid; i < CHUNKL * NST; i += 256) {
    int s = i >> 6, j = i & 63;
    Cs[s * 64 + j] = xbc[(size_t)(t0 + s) * CONV_DIM + D_INNER + NQK * D_STATE + g * 64 + j];
  }
  for (int i = tid; i < HP * NST; i += 256)
    stT[i] = (bf16)states_used[(size_t)bch * (HP * NST) + i];
  __syncthreads();

  int w = tid >> 5, lane = tid & 31;
  int rsub = (lane >> 4) * 8, csub = lane & 15;
  v8f acc[4];
#pragma unroll
  for (int n = 0; n < 4; ++n) acc[n] = zero_v8f();
  for (int k0 = 0; k0 < NST; k0 += 32) {
    v16bf a = load_a_frag(Cs + (w * 16) * NST + k0, NST, lane);
#pragma unroll
    for (int n = 0; n < 4; ++n) {
      v16bf bb = load_bt_frag(stT + (n * 16) * NST + k0, NST, lane);
      acc[n] = wmma_bf16(a, bb, acc[n]);
    }
  }
#pragma unroll
  for (int n = 0; n < 4; ++n)
#pragma unroll
    for (int r = 0; r < 8; ++r) {
      int i = w * 16 + rsub + r;
      int p = n * 16 + csub;
      size_t o = (size_t)(t0 + i) * D_INNER + h * 64 + p;
      Yg[o] += acc[n][r] * __expf(Ac[i]);
    }
}

// ---------------- gating: (Y + D*x/dt) * silu(z + z_bias) -> bf16 ----------------
__global__ void gate_kernel(const float* __restrict__ Yg, const bf16* __restrict__ xbc,
                            const float* __restrict__ dt, const float* __restrict__ xbcza,
                            const float* __restrict__ zb, const float* __restrict__ Dp,
                            bf16* __restrict__ yg) {
  int idx = blockIdx.x * 256 + threadIdx.x;
  if (idx >= TOK * D_INNER) return;
  int t = idx >> 12;
  int d = idx & (D_INNER - 1);
  int h = d >> 6;
  float x = (float)xbc[(size_t)t * CONV_DIM + d];
  float dtv = dt[t * NVH + h];
  float yf = Yg[idx] + Dp[h] * (x / dtv);
  float z = xbcza[(size_t)t * PROJ_DIM + CONV_DIM + d] + zb[d];
  float sz = z / (1.f + __expf(-z));
  yg[idx] = (bf16)(yf * sz);
}

// ---------------- launch ----------------
extern "C" void kernel_launch(void* const* d_in, const int* in_sizes, int n_in,
                              void* d_out, int out_size, void* d_ws, size_t ws_size,
                              hipStream_t stream) {
  const float* input = (const float*)d_in[0];
  const float* inW   = (const float*)d_in[1];
  const float* convW = (const float*)d_in[2];
  const float* convB = (const float*)d_in[3];
  const float* zB    = (const float*)d_in[4];
  const float* Dp    = (const float*)d_in[5];
  const float* outW  = (const float*)d_in[6];
  float* out = (float*)d_out;

  char* ws = (char*)d_ws;
  size_t off = 0;
  auto alloc = [&](size_t bytes) -> void* {
    void* p = (void*)(ws + off);
    off += (bytes + 255) & ~(size_t)255;
    return p;
  };
  bf16*  inp_bf  = (bf16*) alloc((size_t)TOK * D_MODEL * 2);
  bf16*  inW_bf  = (bf16*) alloc((size_t)PROJ_DIM * D_MODEL * 2);
  bf16*  outW_bf = (bf16*) alloc((size_t)D_MODEL * D_INNER * 2);
  float* xbcza   = (float*)alloc((size_t)TOK * PROJ_DIM * 4);
  bf16*  xbc     = (bf16*) alloc((size_t)TOK * CONV_DIM * 2);
  float* dtb     = (float*)alloc((size_t)TOK * NVH * 4);
  float* acg     = (float*)alloc((size_t)BATCH * NCH * NVH * CHUNKL * 4);
  float* states  = (float*)alloc((size_t)BATCH * NCH * NVH * HP * NST * 4);
  float* states2 = (float*)alloc((size_t)BATCH * NCH * NVH * HP * NST * 4);
  float* Yg      = (float*)alloc((size_t)TOK * D_INNER * 4);
  bf16*  yg      = (bf16*) alloc((size_t)TOK * D_INNER * 2);

  int n1 = TOK * D_MODEL;
  cvt_f32_bf16<<<(n1 + 255) / 256, 256, 0, stream>>>(input, inp_bf, n1);
  int n2 = PROJ_DIM * D_MODEL;
  cvt_f32_bf16<<<(n2 + 255) / 256, 256, 0, stream>>>(inW, inW_bf, n2);
  int n3 = D_MODEL * D_INNER;
  cvt_f32_bf16<<<(n3 + 255) / 256, 256, 0, stream>>>(outW, outW_bf, n3);

  gemm_bf16_wmma<<<dim3(TOK / 128, PROJ_DIM / 32), 32, 0, stream>>>(
      inp_bf, inW_bf, xbcza, TOK, PROJ_DIM, D_MODEL);

  int ncv = TOK * CONV_DIM;
  conv_silu_kernel<<<(ncv + 255) / 256, 256, 0, stream>>>(xbcza, convW, convB, xbc);
  int ndt = TOK * NVH;
  dt_kernel<<<(ndt + 255) / 256, 256, 0, stream>>>(xbcza, dtb);

  ssd_chunk_kernel<<<BATCH * NCH * NVH, 256, 0, stream>>>(xbc, dtb, Yg, states, acg);
  chunk_scan_kernel<<<BATCH * NVH, 256, 0, stream>>>(states, acg, states2);
  y_off_kernel<<<BATCH * NCH * NVH, 256, 0, stream>>>(xbc, states2, acg, Yg);

  int ng = TOK * D_INNER;
  gate_kernel<<<(ng + 255) / 256, 256, 0, stream>>>(Yg, xbc, dtb, xbcza, zB, Dp, yg);

  gemm_bf16_wmma<<<dim3(TOK / 128, D_MODEL / 32), 32, 0, stream>>>(
      yg, outW_bf, out, TOK, D_MODEL, D_INNER);
}
